// LSTM_Action_Classifier_Hoidini_69209103008424
// MI455X (gfx1250) — matrix operand order repaired and checked by reading, not compiled, not run on hardware
//
#include <hip/hip_runtime.h>

typedef __attribute__((ext_vector_type(16))) _Float16 v16h;
typedef __attribute__((ext_vector_type(8)))  _Float16 v8h;
typedef __attribute__((ext_vector_type(8)))  float    v8f;
typedef __attribute__((ext_vector_type(4)))  float    v4f;

constexpr int Bc = 256;   // batch
constexpr int Tc = 512;   // time
constexpr int Fc = 263;   // input features
constexpr int Hc = 128;   // hidden
constexpr int Lc = 29;    // labels
constexpr int G4 = 4 * Hc;      // 512 gates
constexpr int WPAD = 136;       // padded LDS row (halves): 272B, 16B aligned, staggers banks

// ---------------------------------------------------------------------------
// Branch-free transcendentals (keep EXEC untouched inside the recurrence).
// ---------------------------------------------------------------------------
__device__ __forceinline__ float sigmoid_fast(float x)
{
    return __fdividef(1.f, 1.f + __expf(-x));     // v_exp + v_rcp, no branches
}

__device__ __forceinline__ float tanh_fast(float x)
{
#if __has_builtin(__builtin_amdgcn_tanhf)
    return __builtin_amdgcn_tanhf(x);             // native v_tanh_f32 on gfx1250
#else
    float e = __expf(-2.f * __builtin_fabsf(x));  // e in (0,1], no overflow
    float r = __fdividef(1.f - e, 1.f + e);
    return __builtin_copysignf(r, x);
#endif
}

// ---------------------------------------------------------------------------
// Input projection GEMM: out[t][b][g] = sum_k A[b*T+t][k] * W[g][k] + bih[g]+bhh[g]
// A is f32 (layer0, K=263) or f16 (layer1, K=128). Row stride of A and W == K.
// Block = 256 threads (8 waves), covers 64 rows x 512 cols.
// Wave decomposition: mgrp(2) x ngrp(4); each wave: 2 M-tiles x 8 N-tiles.
// ---------------------------------------------------------------------------
template <bool AF16, int K>
__global__ void __launch_bounds__(256)
gemm_inproj(const void* __restrict__ Aptr,
            const float* __restrict__ W,
            const float* __restrict__ bih,
            const float* __restrict__ bhh,
            float* __restrict__ out)
{
    const int wv   = threadIdx.x >> 5;
    const int lane = threadIdx.x & 31;
    const int half = lane >> 4;
    const int l16  = lane & 15;
    const int mgrp = wv & 1;
    const int ngrp = wv >> 1;
    const int row0  = blockIdx.x * 64 + mgrp * 32;
    const int ncol0 = ngrp * 128;

    v8f acc[2][8] = {};

    constexpr int KC = (K + 31) >> 5;
#pragma unroll
    for (int kc = 0; kc < KC; ++kc) {
        const int k0 = kc << 5;

        // ---- A fragments (16x32 f16): lane(m=l16), elem i -> k = k0 + 16*half + i
        v16h a[2];
#pragma unroll
        for (int mt = 0; mt < 2; ++mt) {
            const int row = row0 + mt * 16 + l16;
            if constexpr (AF16) {
                const _Float16* Ar = (const _Float16*)Aptr + (size_t)row * K;
                v8h lo = *(const v8h*)(Ar + k0 + 16 * half);          // 16B aligned
                v8h hi = *(const v8h*)(Ar + k0 + 16 * half + 8);
#pragma unroll
                for (int i = 0; i < 8; ++i) { a[mt][i] = lo[i]; a[mt][i + 8] = hi[i]; }
            } else {
                const float* Ar = (const float*)Aptr + (size_t)row * K;
#pragma unroll
                for (int i = 0; i < 16; ++i) {
                    int k = k0 + 16 * half + i;
                    a[mt][i] = (k < K) ? (_Float16)Ar[k] : (_Float16)0.f;
                }
            }
        }

        // ---- B fragments (32x16 f16): lane(n=l16), elem i -> k = k0 + 16*half + i
#pragma unroll
        for (int nt = 0; nt < 8; ++nt) {
            const int n = ncol0 + nt * 16 + l16;
            const float* Wr = W + (size_t)n * K;
            v16h bf;
            if constexpr (AF16) {           // K=128: rows 512B aligned, chunk 64B aligned
                const float* p = Wr + k0 + 16 * half;
                v4f q0 = ((const v4f*)p)[0];
                v4f q1 = ((const v4f*)p)[1];
                v4f q2 = ((const v4f*)p)[2];
                v4f q3 = ((const v4f*)p)[3];
#pragma unroll
                for (int i = 0; i < 4; ++i) {
                    bf[i]      = (_Float16)q0[i];
                    bf[i + 4]  = (_Float16)q1[i];
                    bf[i + 8]  = (_Float16)q2[i];
                    bf[i + 12] = (_Float16)q3[i];
                }
            } else {                        // K=263: odd stride, stay on dword loads
#pragma unroll
                for (int i = 0; i < 16; ++i) {
                    int k = k0 + 16 * half + i;
                    bf[i] = (k < K) ? (_Float16)Wr[k] : (_Float16)0.f;
                }
            }
#pragma unroll
            for (int mt = 0; mt < 2; ++mt) {
                acc[mt][nt] = __builtin_amdgcn_wmma_f32_16x16x32_f16(
                    false, a[mt], false, bf, (short)0, acc[mt][nt], false, false);
            }
        }
    }

    // ---- epilogue: D elem r -> m = r + 8*half ; out layout [t][b][g]
#pragma unroll
    for (int nt = 0; nt < 8; ++nt) {
        const int n = ncol0 + nt * 16 + l16;
        const float bias = bih[n] + bhh[n];
#pragma unroll
        for (int mt = 0; mt < 2; ++mt) {
            const int rowb = row0 + mt * 16;
#pragma unroll
            for (int r = 0; r < 8; ++r) {
                const int row = rowb + r + 8 * half;
                const int b = row >> 9;        // row / T (T=512)
                const int t = row & 511;       // row % T
                out[((size_t)t * Bc + b) * G4 + n] = acc[mt][nt][r] + bias;
            }
        }
    }
}

// ---------------------------------------------------------------------------
// LSTM recurrence. grid = 16 WGs (16 batch rows each), block = 256 (8 waves).
// W_hh B-fragments live in REGISTERS (16 x v16h = 128 VGPRs per wave, constant
// across all timesteps) -> steady-state step is 8 ds_load_b128 (h fragments),
// one wait, 16 register-resident WMMAs, branch-free elementwise, h store.
// xproj for step t+1 (scalar-clamped, no control flow) is loaded directly into
// the accumulators after the elementwise consumes them, hiding global latency
// behind the store/barrier/A-load phase. Wave w owns h-columns [16w,16w+16)
// and gate types j=0..3 (i,f,g,o) for them, so cell state c[8] is register-local.
// ---------------------------------------------------------------------------
__global__ void __launch_bounds__(256, 1)
lstm_recurrent(const float* __restrict__ xproj,
               const float* __restrict__ Whh,
               _Float16* __restrict__ hout)
{
    __shared__ _Float16 h_lds[16 * WPAD];     // 16*136*2 = 4352 B

    const int tid  = threadIdx.x;
    const int wv   = tid >> 5;
    const int lane = tid & 31;
    const int half = lane >> 4;
    const int l16  = lane & 15;
    const int b0   = blockIdx.x * 16;
    const int hcol = wv * 16 + l16;

    // ---- load this wave's W_hh B-fragments into registers (one time).
    // B-frag (32x16): lane holds col n = j*128 + hcol, elem i -> k = kt*32 + 16*half + i
    v16h breg[4][4];
#pragma unroll
    for (int j = 0; j < 4; ++j) {
        const int n = j * 128 + hcol;
        const float* Wr = Whh + (size_t)n * Hc;
#pragma unroll
        for (int kt = 0; kt < 4; ++kt) {
            const float* p = Wr + kt * 32 + 16 * half;
            v4f q0 = ((const v4f*)p)[0];   // Whh rows are 512B aligned, chunks 64B aligned
            v4f q1 = ((const v4f*)p)[1];
            v4f q2 = ((const v4f*)p)[2];
            v4f q3 = ((const v4f*)p)[3];
#pragma unroll
            for (int i = 0; i < 4; ++i) {
                breg[j][kt][i]      = (_Float16)q0[i];
                breg[j][kt][i + 4]  = (_Float16)q1[i];
                breg[j][kt][i + 8]  = (_Float16)q2[i];
                breg[j][kt][i + 12] = (_Float16)q3[i];
            }
        }
    }

    // ---- zero h(–1)
    for (int idx = tid; idx < 16 * WPAD; idx += 256) h_lds[idx] = (_Float16)0.f;

    float c[8];
#pragma unroll
    for (int r = 0; r < 8; ++r) c[r] = 0.f;

    // ---- accumulators pre-seeded with xproj for t=0
    v8f acc[4];
#pragma unroll
    for (int j = 0; j < 4; ++j)
#pragma unroll
        for (int r = 0; r < 8; ++r)
            acc[j][r] = xproj[(size_t)(b0 + r + 8 * half) * G4 + j * 128 + hcol];

    __syncthreads();

    for (int t = 0; t < Tc; ++t) {
        // ---- A fragments from h(t-1) in LDS: elem i -> k = kt*32 + 16*half + i
        v16h a[4];
#pragma unroll
        for (int kt = 0; kt < 4; ++kt) {
            const _Float16* hp = &h_lds[l16 * WPAD + kt * 32 + 16 * half];
            v8h lo = *(const v8h*)hp;
            v8h hi = *(const v8h*)(hp + 8);
#pragma unroll
            for (int i = 0; i < 8; ++i) { a[kt][i] = lo[i]; a[kt][i + 8] = hi[i]; }
        }
        __syncthreads();   // everyone has read h(t-1); safe to overwrite below

        // ---- gates: 16 register-resident WMMAs, no memory in the chain
#pragma unroll
        for (int j = 0; j < 4; ++j)
#pragma unroll
            for (int kt = 0; kt < 4; ++kt)
                acc[j] = __builtin_amdgcn_wmma_f32_16x16x32_f16(
                    false, a[kt], false, breg[j][kt], (short)0, acc[j], false, false);

        // ---- elementwise cell update (register-local, branch-free)
        float hn[8];
#pragma unroll
        for (int r = 0; r < 8; ++r) {
            float si = sigmoid_fast(acc[0][r]);
            float sf = sigmoid_fast(acc[1][r]);
            float tg = tanh_fast(acc[2][r]);
            float so = sigmoid_fast(acc[3][r]);
            c[r] = sf * c[r] + si * tg;
            hn[r] = so * tanh_fast(c[r]);
        }

        // ---- refill accumulators with next step's xproj, unconditional
        //      (scalar clamp keeps the loop a single basic block; final
        //      iteration re-reads t=Tc-1 into dead registers)
        {
            const int tnext = (t + 1 < Tc) ? (t + 1) : (Tc - 1);   // s_cselect, no branch
#pragma unroll
            for (int j = 0; j < 4; ++j)
#pragma unroll
                for (int r = 0; r < 8; ++r)
                    acc[j][r] = xproj[((size_t)tnext * Bc + (b0 + r + 8 * half)) * G4 + j * 128 + hcol];
        }

        // ---- publish h(t) to LDS + global (f16, layout [b][t][h])
#pragma unroll
        for (int r = 0; r < 8; ++r) {
            const int m = r + 8 * half;
            const _Float16 hv = (_Float16)hn[r];
            h_lds[m * WPAD + hcol] = hv;
            hout[((size_t)(b0 + m) * Tc + t) * Hc + hcol] = hv;
        }
        __syncthreads();
    }
}

// ---------------------------------------------------------------------------
// Head: logits = h2[:, T-1, :] @ Wout^T + bout ; sigmoid ; per-batch BCE partials
// ---------------------------------------------------------------------------
__global__ void __launch_bounds__(32)
head_kernel(const _Float16* __restrict__ h2,
            const float* __restrict__ Wout,
            const float* __restrict__ bout,
            const int* __restrict__ y,
            float* __restrict__ out,
            float* __restrict__ partial)
{
    const int b = blockIdx.x;
    const int l = threadIdx.x;
    float term = 0.f;
    if (l < Lc) {
        const _Float16* hp = h2 + ((size_t)b * Tc + (Tc - 1)) * Hc;
        float accf = 0.f;
#pragma unroll
        for (int k = 0; k < Hc; ++k) accf += (float)hp[k] * Wout[l * Hc + k];
        accf += bout[l];
        float p = sigmoid_fast(accf);
        out[1 + b * Lc + l] = p;
        float yf = (float)y[b * Lc + l];
        const float eps = 1e-12f;
        term = yf * __logf(p + eps) + (1.f - yf) * __logf(1.f - p + eps);
    }
#pragma unroll
    for (int off = 16; off > 0; off >>= 1) term += __shfl_down(term, off, 32);
    if (l == 0) partial[b] = term;
}

__global__ void __launch_bounds__(256)
loss_reduce(const float* __restrict__ partial, float* __restrict__ out)
{
    __shared__ float s[256];
    s[threadIdx.x] = partial[threadIdx.x];
    __syncthreads();
    for (int off = 128; off > 0; off >>= 1) {
        if (threadIdx.x < off) s[threadIdx.x] += s[threadIdx.x + off];
        __syncthreads();
    }
    if (threadIdx.x == 0) out[0] = -s[0] / (float)(Bc * Lc);
}

// ---------------------------------------------------------------------------
extern "C" void kernel_launch(void* const* d_in, const int* in_sizes, int n_in,
                              void* d_out, int out_size, void* d_ws, size_t ws_size,
                              hipStream_t stream)
{
    (void)in_sizes; (void)n_in; (void)out_size; (void)ws_size;

    const float* x    = (const float*)d_in[0];
    const int*   y    = (const int*)  d_in[1];
    const float* Wih0 = (const float*)d_in[2];
    const float* Whh0 = (const float*)d_in[3];
    const float* bih0 = (const float*)d_in[4];
    const float* bhh0 = (const float*)d_in[5];
    const float* Wih1 = (const float*)d_in[6];
    const float* Whh1 = (const float*)d_in[7];
    const float* bih1 = (const float*)d_in[8];
    const float* bhh1 = (const float*)d_in[9];
    const float* Wout = (const float*)d_in[10];
    const float* bout = (const float*)d_in[11];
    float* out = (float*)d_out;

    char* ws = (char*)d_ws;
    const size_t xproj_bytes = (size_t)Tc * Bc * G4 * sizeof(float);     // 268 MB
    const size_t hbuf_bytes  = (size_t)Bc * Tc * Hc * sizeof(_Float16);  // 33.5 MB
    float*     xproj   = (float*)ws;
    _Float16*  hbuf    = (_Float16*)(ws + xproj_bytes);
    float*     partial = (float*)(ws + xproj_bytes + hbuf_bytes);

    const int gemm_grid = (Bc * Tc) / 64;   // 2048
    dim3 blk(256);

    // layer 0
    gemm_inproj<false, Fc><<<gemm_grid, blk, 0, stream>>>(x, Wih0, bih0, bhh0, xproj);
    lstm_recurrent<<<Bc / 16, blk, 0, stream>>>(xproj, Whh0, hbuf);
    // layer 1
    gemm_inproj<true, Hc><<<gemm_grid, blk, 0, stream>>>(hbuf, Wih1, bih1, bhh1, xproj);
    lstm_recurrent<<<Bc / 16, blk, 0, stream>>>(xproj, Whh1, hbuf);
    // head + loss
    head_kernel<<<Bc, 32, 0, stream>>>(hbuf, Wout, bout, y, out, partial);
    loss_reduce<<<1, 256, 0, stream>>>(partial, out);
}
